// LinearTransformerExpert_4054449127668
// MI455X (gfx1250) — matrix operand order repaired and checked by reading, not compiled, not run on hardware
//
#include <hip/hip_runtime.h>

typedef __attribute__((ext_vector_type(16))) __bf16 v16bf;
typedef __attribute__((ext_vector_type(8)))  __bf16 v8bf;
typedef __attribute__((ext_vector_type(8)))  float  v8f;
typedef __attribute__((ext_vector_type(4)))  float  v4f;

#define TILE   128
#define KT     32
#define LDSS   40   // bf16 elements per LDS row (32 data + 8 pad); 80B rows keep 16B alignment
#define KDIM   2048

// Convert 16 consecutive f32 from global to bf16 and store to LDS (two b128 stores).
__device__ __forceinline__ void stage16_f32(const float* __restrict__ gsrc, __bf16* ldst) {
  const v4f* s4 = (const v4f*)gsrc;
  v4f f0 = s4[0], f1 = s4[1], f2 = s4[2], f3 = s4[3];
  v8bf p0, p1;
#pragma unroll
  for (int e = 0; e < 4; ++e) {
    p0[e]     = (__bf16)f0[e];
    p0[e + 4] = (__bf16)f1[e];
    p1[e]     = (__bf16)f2[e];
    p1[e + 4] = (__bf16)f3[e];
  }
  *(v8bf*)ldst       = p0;
  *(v8bf*)(ldst + 8) = p1;
}

// A 16x32 bf16 fragment: lanes 0-15 -> M=l, K {0..7,16..23}; lanes 16-31 -> M=l, K {8..15,24..31}
__device__ __forceinline__ v16bf load_afrag(const __bf16* As, int row, int hf) {
  const int c0 = hf ? 8 : 0;
  const int c1 = hf ? 24 : 16;
  v8bf lo = *(const v8bf*)&As[row * LDSS + c0];
  v8bf hi = *(const v8bf*)&As[row * LDSS + c1];
  v16bf f;
#pragma unroll
  for (int e = 0; e < 8; ++e) { f[e] = lo[e]; f[e + 8] = hi[e]; }
  return f;
}

// B 32x16 bf16 fragment: lane N=l; lanes 0-15 hold K {0..7,16..23}, lanes 16-31 hold K {8..15,24..31}
__device__ __forceinline__ v16bf load_bfrag(const __bf16* Bs, int col, int hf) {
  const int kb = hf * 8;
  v8bf lo = *(const v8bf*)&Bs[col * LDSS + kb];
  v8bf hi = *(const v8bf*)&Bs[col * LDSS + kb + 16];
  v16bf f;
#pragma unroll
  for (int e = 0; e < 8; ++e) { f[e] = lo[e]; f[e + 8] = hi[e]; }
  return f;
}

// Dual-output GEMM: Y1 = A @ W1^T, Y2 = A @ W2^T. A-tile staged once, used twice.
// MODE 0 (KV): both outputs bf16.  MODE 1 (QG): O1 bf16, O2 = sigmoid(acc + bias[col]) f32.
template <int MODE>
__global__ __launch_bounds__(256) void gemm_dual_wmma(
    const float* __restrict__ A,    // [M, 2048] f32
    const float* __restrict__ W1,   // [2048, 2048] f32
    const float* __restrict__ W2,   // [2048, 2048] f32
    void* __restrict__ O1,
    void* __restrict__ O2,
    const float* __restrict__ bias) {
  __shared__ __bf16 As [TILE * LDSS];
  __shared__ __bf16 B1s[TILE * LDSS];
  __shared__ __bf16 B2s[TILE * LDSS];

  const int bm = blockIdx.y * TILE;
  const int bn = blockIdx.x * TILE;
  const int t    = threadIdx.x;
  const int lane = t & 31;
  const int wave = t >> 5;
  const int wm = (wave & 3) * 32;   // 4 waves tile M
  const int wn = (wave >> 2) * 64;  // 2 waves tile N
  const int l  = lane & 15;
  const int hf = lane >> 4;

  v8f acc1[2][4], acc2[2][4];
#pragma unroll
  for (int i = 0; i < 2; ++i)
#pragma unroll
    for (int j = 0; j < 4; ++j) {
      v8f z = {0.f, 0.f, 0.f, 0.f, 0.f, 0.f, 0.f, 0.f};
      acc1[i][j] = z;
      acc2[i][j] = z;
    }

  const int lr = t >> 1;         // 0..127: tile row for cooperative staging
  const int lc = (t & 1) * 16;   // 0 or 16: K-chunk base

  const float* aRow  = A  + (size_t)(bm + lr) * KDIM + lc;
  const float* w1Row = W1 + (size_t)(bn + lr) * KDIM + lc;
  const float* w2Row = W2 + (size_t)(bn + lr) * KDIM + lc;

  for (int kk = 0; kk < KDIM; kk += KT) {
    stage16_f32(aRow  + kk, &As [lr * LDSS + lc]);
    stage16_f32(w1Row + kk, &B1s[lr * LDSS + lc]);
    stage16_f32(w2Row + kk, &B2s[lr * LDSS + lc]);
    if (kk + KT < KDIM) {
      __builtin_prefetch(aRow  + kk + KT, 0, 3);
      __builtin_prefetch(w1Row + kk + KT, 0, 3);
      __builtin_prefetch(w2Row + kk + KT, 0, 3);
    }
    __syncthreads();

    v16bf afrag[2];
#pragma unroll
    for (int i = 0; i < 2; ++i) afrag[i] = load_afrag(As, wm + i * 16 + l, hf);

    {
      v16bf bfrag[4];
#pragma unroll
      for (int j = 0; j < 4; ++j) bfrag[j] = load_bfrag(B1s, wn + j * 16 + l, hf);
#pragma unroll
      for (int i = 0; i < 2; ++i)
#pragma unroll
        for (int j = 0; j < 4; ++j)
          acc1[i][j] = __builtin_amdgcn_wmma_f32_16x16x32_bf16(
              false, afrag[i], false, bfrag[j], (short)0, acc1[i][j], false, false);
    }
    {
      v16bf bfrag[4];
#pragma unroll
      for (int j = 0; j < 4; ++j) bfrag[j] = load_bfrag(B2s, wn + j * 16 + l, hf);
#pragma unroll
      for (int i = 0; i < 2; ++i)
#pragma unroll
        for (int j = 0; j < 4; ++j)
          acc2[i][j] = __builtin_amdgcn_wmma_f32_16x16x32_bf16(
              false, afrag[i], false, bfrag[j], (short)0, acc2[i][j], false, false);
    }
    __syncthreads();
  }

  // C/D f32 16x16 layout: VGPR v, lane: row = v + 8*hf, col = l
#pragma unroll
  for (int i = 0; i < 2; ++i)
#pragma unroll
    for (int j = 0; j < 4; ++j)
#pragma unroll
      for (int v = 0; v < 8; ++v) {
        const size_t row = (size_t)(bm + wm + i * 16 + v + hf * 8);
        const size_t col = (size_t)(bn + wn + j * 16 + l);
        const size_t idx = row * (size_t)KDIM + col;
        const float x1 = acc1[i][j][v];
        const float x2 = acc2[i][j][v];
        ((__bf16*)O1)[idx] = (__bf16)x1;
        if (MODE == 0) {
          ((__bf16*)O2)[idx] = (__bf16)x2;
        } else {
          ((float*)O2)[idx] = 1.0f / (1.0f + __expf(-(x2 + bias[col])));
        }
      }
}

// Output projection: out = (Abf16 @ Wo^T) * g, f32 result.
// A-tile is already bf16 in memory -> staged with async global->LDS copies (ASYNCcnt path).
__global__ __launch_bounds__(256) void gemm_out_wmma(
    const __bf16* __restrict__ A,   // [8192, 2048] bf16 (attention output)
    const float* __restrict__ W,    // [2048, 2048] f32
    const float* __restrict__ gmul, // [8192, 2048] f32 gate
    float* __restrict__ Out) {
  __shared__ __bf16 As[TILE * LDSS];
  __shared__ __bf16 Bs[TILE * LDSS];

  const int bm = blockIdx.y * TILE;
  const int bn = blockIdx.x * TILE;
  const int t    = threadIdx.x;
  const int lane = t & 31;
  const int wave = t >> 5;
  const int wm = (wave & 3) * 32;
  const int wn = (wave >> 2) * 64;
  const int l  = lane & 15;
  const int hf = lane >> 4;

  v8f acc[2][4];
#pragma unroll
  for (int i = 0; i < 2; ++i)
#pragma unroll
    for (int j = 0; j < 4; ++j) {
      v8f z = {0.f, 0.f, 0.f, 0.f, 0.f, 0.f, 0.f, 0.f};
      acc[i][j] = z;
    }

  const int lr = t >> 1;
  const int lc = (t & 1) * 16;
  const __bf16* aRow = A + (size_t)(bm + lr) * KDIM + lc;
  const float*  wRow = W + (size_t)(bn + lr) * KDIM + lc;

  // LDS byte address for this thread's A-tile slot: low 32 bits of the generic
  // pointer carry the wave-relative LDS offset (flat->LDS truncates to addr[31:0]).
  const unsigned int ldsA = (unsigned int)(unsigned long long)(&As[lr * LDSS + lc]);

  for (int kk = 0; kk < KDIM; kk += KT) {
    {
      // Async raw copy of 32 bytes of bf16 A directly into LDS, no VGPR round-trip.
      // INST_OFFSET applies to both global and LDS addresses, so offset:16 covers
      // the second 16-byte chunk with the same address operands.
      const __bf16* gptr = aRow + kk;
      asm volatile("global_load_async_to_lds_b128 %0, %1, off"
                   :: "v"(ldsA), "v"(gptr) : "memory");
      asm volatile("global_load_async_to_lds_b128 %0, %1, off offset:16"
                   :: "v"(ldsA), "v"(gptr) : "memory");
    }
    stage16_f32(wRow + kk, &Bs[lr * LDSS + lc]);
    if (kk + KT < KDIM) {
      __builtin_prefetch(aRow + kk + KT, 0, 3);
      __builtin_prefetch(wRow + kk + KT, 0, 3);
    }
    // Wave-local completion of async LDS writes, then workgroup visibility barrier.
    asm volatile("s_wait_asynccnt 0x0" ::: "memory");
    __syncthreads();

    v16bf afrag[2];
#pragma unroll
    for (int i = 0; i < 2; ++i) afrag[i] = load_afrag(As, wm + i * 16 + l, hf);
    v16bf bfrag[4];
#pragma unroll
    for (int j = 0; j < 4; ++j) bfrag[j] = load_bfrag(Bs, wn + j * 16 + l, hf);

#pragma unroll
    for (int i = 0; i < 2; ++i)
#pragma unroll
      for (int j = 0; j < 4; ++j)
        acc[i][j] = __builtin_amdgcn_wmma_f32_16x16x32_bf16(
            false, afrag[i], false, bfrag[j], (short)0, acc[i][j], false, false);
    __syncthreads();
  }

#pragma unroll
  for (int i = 0; i < 2; ++i)
#pragma unroll
    for (int j = 0; j < 4; ++j)
#pragma unroll
      for (int v = 0; v < 8; ++v) {
        const size_t row = (size_t)(bm + wm + i * 16 + v + hf * 8);
        const size_t col = (size_t)(bn + wn + j * 16 + l);
        const size_t idx = row * (size_t)KDIM + col;
        Out[idx] = acc[i][j][v] * gmul[idx];
      }
}

// One thread per (n, h): scores over P=8, softmax, weighted sum of v. D=128.
__global__ __launch_bounds__(256) void attn_kernel(
    const __bf16* __restrict__ q,    // [8192, 2048]
    const __bf16* __restrict__ k,    // [65536, 2048] (rows n*8+p)
    const __bf16* __restrict__ v,    // [65536, 2048]
    __bf16* __restrict__ o)          // [8192, 2048]
{
  const int gid = blockIdx.x * 256 + threadIdx.x;  // 0 .. 131071
  const int n = gid >> 4;
  const int h = gid & 15;
  const __bf16* qp = q + (size_t)n * 2048 + h * 128;
  const __bf16* kb = k + (size_t)n * 8 * 2048 + h * 128;
  const __bf16* vb = v + (size_t)n * 8 * 2048 + h * 128;

  float s[8];
#pragma unroll
  for (int p = 0; p < 8; ++p) {
    const v8bf* kp = (const v8bf*)(kb + (size_t)p * 2048);
    const v8bf* qq = (const v8bf*)qp;
    float acc = 0.f;
    for (int c = 0; c < 16; ++c) {
      v8bf a = qq[c], b = kp[c];
#pragma unroll
      for (int e = 0; e < 8; ++e) acc += (float)a[e] * (float)b[e];
    }
    s[p] = acc * 0.08838834764831845f;  // 1/sqrt(128)
  }
  float m = s[0];
#pragma unroll
  for (int p = 1; p < 8; ++p) m = fmaxf(m, s[p]);
  float sum = 0.f;
#pragma unroll
  for (int p = 0; p < 8; ++p) { s[p] = __expf(s[p] - m); sum += s[p]; }
  const float inv = 1.0f / sum;
#pragma unroll
  for (int p = 0; p < 8; ++p) s[p] *= inv;

  __bf16* op = o + (size_t)n * 2048 + h * 128;
  for (int c = 0; c < 16; ++c) {
    float a8[8] = {0.f, 0.f, 0.f, 0.f, 0.f, 0.f, 0.f, 0.f};
#pragma unroll
    for (int p = 0; p < 8; ++p) {
      v8bf vv = *(const v8bf*)(vb + (size_t)p * 2048 + c * 8);
#pragma unroll
      for (int e = 0; e < 8; ++e) a8[e] += s[p] * (float)vv[e];
    }
    v8bf pk;
#pragma unroll
    for (int e = 0; e < 8; ++e) pk[e] = (__bf16)a8[e];
    *(v8bf*)(op + c * 8) = pk;
  }
}

extern "C" void kernel_launch(void* const* d_in, const int* in_sizes, int n_in,
                              void* d_out, int out_size, void* d_ws, size_t ws_size,
                              hipStream_t stream) {
  (void)in_sizes; (void)n_in; (void)out_size; (void)ws_size;
  const float* x      = (const float*)d_in[0];  // [8192, 2048]
  const float* prefix = (const float*)d_in[1];  // [8192, 8, 2048] -> [65536, 2048]
  const float* Wq     = (const float*)d_in[2];
  const float* Wk     = (const float*)d_in[3];
  const float* Wv     = (const float*)d_in[4];
  const float* Wo     = (const float*)d_in[5];
  const float* Wg     = (const float*)d_in[6];
  const float* bg     = (const float*)d_in[7];
  float* out = (float*)d_out;

  char* ws = (char*)d_ws;
  const size_t SZ_Q  = (size_t)8192 * 2048 * 2;   // 33.5 MB bf16
  const size_t SZ_KV = (size_t)65536 * 2048 * 2;  // 268 MB bf16
  __bf16* qb = (__bf16*)(ws);
  __bf16* kb = (__bf16*)(ws + SZ_Q);
  __bf16* vb = (__bf16*)(ws + SZ_Q + SZ_KV);
  __bf16* ab = (__bf16*)(ws + SZ_Q + SZ_KV + SZ_KV);
  float*  g  = (float*) (ws + SZ_Q + SZ_KV + SZ_KV + SZ_Q);

  dim3 blk(256);
  dim3 gQ(16, 64);    // 2048/128 x 8192/128
  dim3 gKV(16, 512);  // 2048/128 x 65536/128

  // Fused Q+G projections (read x once): q bf16, g = sigmoid(x@Wg^T + bg) f32
  gemm_dual_wmma<1><<<gQ,  blk, 0, stream>>>(x,      Wq, Wg, qb, g,  bg);
  // Fused K+V projections (read prefix once): k, v bf16
  gemm_dual_wmma<0><<<gKV, blk, 0, stream>>>(prefix, Wk, Wv, kb, vb, nullptr);
  // Softmax attention over P=8
  attn_kernel<<<dim3(512), blk, 0, stream>>>(qb, kb, vb, ab);
  // Output projection with gating epilogue
  gemm_out_wmma<<<gQ, blk, 0, stream>>>(ab, Wo, g, out);
}
